// MemoryEfficientAttention_47399259079167
// MI455X (gfx1250) — matrix operand order repaired and checked by reading, not compiled, not run on hardware
//
#include <hip/hip_runtime.h>

typedef __attribute__((ext_vector_type(16))) __bf16 bf16x16;
typedef __attribute__((ext_vector_type(2)))  __bf16 bf16x2;
typedef __attribute__((ext_vector_type(8)))  float  f32x8;

struct U2 { unsigned x, y; };
struct U4 { unsigned x, y, z, w; };

union AFrag {
  bf16x16 v;
  unsigned u[8];
  unsigned short h[16];
};

static constexpr int B_ = 4, S_ = 4096, D_ = 256;
static constexpr int QT = 64;   // query rows per workgroup
static constexpr int NW = 4;    // waves per workgroup
static constexpr int KT = 32;   // keys per KV tile
// (1/sqrt(256)) * log2(e): do softmax in base-2
static constexpr float KSCALE = 0.09016844005555897f;

// Pack two f32 into two bf16 (one dword). Prefer hardware v_cvt_pk_bf16_f32.
__device__ __forceinline__ unsigned pk_bf16(float lo, float hi) {
#if __has_builtin(__builtin_amdgcn_cvt_pk_bf16_f32)
  bf16x2 r = __builtin_amdgcn_cvt_pk_bf16_f32(lo, hi);
  return __builtin_bit_cast(unsigned, r);
#else
  // truncation: v_lshrrev_b32 + v_and_or_b32 (2 VALU for 2 elements)
  return (__float_as_uint(lo) >> 16) | (__float_as_uint(hi) & 0xffff0000u);
#endif
}

// Single f32 -> bf16 (truncation, 1 VALU op)
__device__ __forceinline__ unsigned short f2bf1(float f) {
  return (unsigned short)(__float_as_uint(f) >> 16);
}

__global__ __launch_bounds__(128)
void fa_fwd_bf16_wmma(const float* __restrict__ Qg, const float* __restrict__ Kg,
                      const float* __restrict__ Vg, float* __restrict__ Og) {
  __shared__ __align__(16) unsigned short Ks[KT][D_];     // K tile, natural [key][d]
  __shared__ __align__(16) unsigned short Vt[D_][KT];     // V tile, transposed [d][key]
  __shared__ __align__(16) unsigned short Ps[NW][16][KT]; // per-wave P relayout

  const int tid  = threadIdx.x;
  const int lane = tid & 31;
  const int w    = tid >> 5;
  const int half = lane >> 4;   // lane-half select (wave32 WMMA layouts)
  const int ln16 = lane & 15;

  const int b     = blockIdx.x / (S_ / QT);
  const int qbase = (blockIdx.x % (S_ / QT)) * QT;

  // ---- preload this wave's 16x256 Q strip as bf16 A-fragments ----
  AFrag qf[8];
  {
    const int qrow = qbase + 16 * w + ln16;              // A: lane = row
    const float* qp = Qg + ((size_t)b * S_ + qrow) * D_;
    #pragma unroll
    for (int c = 0; c < 8; ++c) {
      const int d0 = 32 * c + 8 * half;                  // h[0..7]
      float4 x0 = *(const float4*)(qp + d0);
      float4 x1 = *(const float4*)(qp + d0 + 4);
      float4 y0 = *(const float4*)(qp + d0 + 16);        // h[8..15]
      float4 y1 = *(const float4*)(qp + d0 + 20);
      qf[c].u[0] = pk_bf16(x0.x, x0.y);
      qf[c].u[1] = pk_bf16(x0.z, x0.w);
      qf[c].u[2] = pk_bf16(x1.x, x1.y);
      qf[c].u[3] = pk_bf16(x1.z, x1.w);
      qf[c].u[4] = pk_bf16(y0.x, y0.y);
      qf[c].u[5] = pk_bf16(y0.z, y0.w);
      qf[c].u[6] = pk_bf16(y1.x, y1.y);
      qf[c].u[7] = pk_bf16(y1.z, y1.w);
    }
  }

  f32x8 acc[16];
  #pragma unroll
  for (int t = 0; t < 16; ++t) acc[t] = (f32x8){0,0,0,0,0,0,0,0};
  float mrun[8], lrun[8];
  #pragma unroll
  for (int r = 0; r < 8; ++r) { mrun[r] = -3.0e38f; lrun[r] = 0.0f; }

  const int ntiles = S_ / KT; // 128

  for (int kt = 0; kt < ntiles; ++kt) {
    const int kbase = kt * KT;

    // ---- cooperative stage of K (natural) / V (transposed) as bf16 ----
    #pragma unroll 4
    for (int j = 0; j < 16; ++j) {
      const int i    = tid + 128 * j;        // float4 index in 32x256 tile
      const int elem = i * 4;
      const int key  = elem >> 8;
      const int d    = elem & 255;
      const size_t goff = ((size_t)b * S_ + kbase + key) * D_ + d;
      float4 kv = *(const float4*)(Kg + goff);
      U2 kp2;
      kp2.x = pk_bf16(kv.x, kv.y);
      kp2.y = pk_bf16(kv.z, kv.w);
      *(U2*)&Ks[key][d] = kp2;
      float4 vv = *(const float4*)(Vg + goff);
      Vt[d + 0][key] = f2bf1(vv.x);
      Vt[d + 1][key] = f2bf1(vv.y);
      Vt[d + 2][key] = f2bf1(vv.z);
      Vt[d + 3][key] = f2bf1(vv.w);
    }
    if (kt + 1 < ntiles) {  // gfx1250 global_prefetch_b8 for next tile
      const size_t noff = ((size_t)b * S_ + kbase + KT) * D_ + ((tid * 64) & (KT * D_ - 1));
      __builtin_prefetch(Kg + noff, 0, 1);
      __builtin_prefetch(Vg + noff, 0, 1);
    }
    __syncthreads();

    // ---- S = Q * K^T : 16 rows x 32 keys, K-dim 256 in 8 chunks ----
    f32x8 s0 = (f32x8){0,0,0,0,0,0,0,0};
    f32x8 s1 = (f32x8){0,0,0,0,0,0,0,0};
    #pragma unroll
    for (int c = 0; c < 8; ++c) {
      AFrag b0, b1;  // B: lane = key(N), d = 8*half+[0..7] and 16+8*half+[0..7]
      const unsigned short* k0 = &Ks[ln16][32 * c + 8 * half];
      const unsigned short* k1 = &Ks[16 + ln16][32 * c + 8 * half];
      *(U4*)&b0.u[0] = *(const U4*)(k0);
      *(U4*)&b0.u[4] = *(const U4*)(k0 + 16);
      *(U4*)&b1.u[0] = *(const U4*)(k1);
      *(U4*)&b1.u[4] = *(const U4*)(k1 + 16);
      s0 = __builtin_amdgcn_wmma_f32_16x16x32_bf16(false, qf[c].v, false, b0.v,
                                                   (short)0, s0, false, false);
      s1 = __builtin_amdgcn_wmma_f32_16x16x32_bf16(false, qf[c].v, false, b1.v,
                                                   (short)0, s1, false, false);
    }

    // ---- online softmax in base-2 over this 32-key slab ----
    float rmax[8], p0[8], p1[8], arow[8], rsum[8];
    #pragma unroll
    for (int r = 0; r < 8; ++r) rmax[r] = fmaxf(s0[r], s1[r]);
    #pragma unroll
    for (int mm = 1; mm <= 8; mm <<= 1) {
      #pragma unroll
      for (int r = 0; r < 8; ++r)
        rmax[r] = fmaxf(rmax[r], __shfl_xor(rmax[r], mm, 32));
    }
    #pragma unroll
    for (int r = 0; r < 8; ++r) {
      const float mn = fmaxf(mrun[r], rmax[r] * KSCALE);
      arow[r] = exp2f(mrun[r] - mn);
      mrun[r] = mn;
      p0[r] = exp2f(s0[r] * KSCALE - mn);
      p1[r] = exp2f(s1[r] * KSCALE - mn);
      rsum[r] = p0[r] + p1[r];
    }
    #pragma unroll
    for (int mm = 1; mm <= 8; mm <<= 1) {
      #pragma unroll
      for (int r = 0; r < 8; ++r)
        rsum[r] += __shfl_xor(rsum[r], mm, 32);
    }
    #pragma unroll
    for (int r = 0; r < 8; ++r) lrun[r] = lrun[r] * arow[r] + rsum[r];
    #pragma unroll
    for (int t = 0; t < 16; ++t) {
      #pragma unroll
      for (int r = 0; r < 8; ++r) acc[t][r] *= arow[r];
    }

    // ---- relayout P (C-format -> A-format) through per-wave LDS ----
    #pragma unroll
    for (int r = 0; r < 8; ++r) {
      Ps[w][r + 8 * half][ln16]      = f2bf1(p0[r]);   // keys 0..15
      Ps[w][r + 8 * half][16 + ln16] = f2bf1(p1[r]);   // keys 16..31
    }
    asm volatile("" ::: "memory"); // DS ops of one wave execute in order
    AFrag pa;
    {
      const unsigned short* pr = &Ps[w][ln16][0];
      *(U4*)&pa.u[0] = *(const U4*)(pr + 8 * half);
      *(U4*)&pa.u[4] = *(const U4*)(pr + 16 + 8 * half);
    }

    // ---- O += P * V : 16 d-column tiles ----
    #pragma unroll
    for (int t = 0; t < 16; ++t) {
      AFrag bv;  // B: lane = d-col(N), keys 8*half+[0..7] and 16+8*half+[0..7]
      const unsigned short* vr = &Vt[16 * t + ln16][8 * half];
      *(U4*)&bv.u[0] = *(const U4*)(vr);
      *(U4*)&bv.u[4] = *(const U4*)(vr + 16);
      acc[t] = __builtin_amdgcn_wmma_f32_16x16x32_bf16(false, pa.v, false, bv.v,
                                                       (short)0, acc[t], false, false);
    }
    __syncthreads();
  }

  // ---- normalize by the softmax denominator and store fp32 output ----
  #pragma unroll
  for (int r = 0; r < 8; ++r) {
    const float inv = 1.0f / lrun[r];
    const int row = qbase + 16 * w + r + 8 * half;     // C layout row mapping
    float* op = Og + ((size_t)b * S_ + row) * D_ + ln16;
    #pragma unroll
    for (int t = 0; t < 16; ++t) op[16 * t] = acc[t][r] * inv;
  }
}

extern "C" void kernel_launch(void* const* d_in, const int* in_sizes, int n_in,
                              void* d_out, int out_size, void* d_ws, size_t ws_size,
                              hipStream_t stream) {
  (void)in_sizes; (void)n_in; (void)out_size; (void)d_ws; (void)ws_size;
  const float* Q = (const float*)d_in[0];
  const float* K = (const float*)d_in[1];
  const float* V = (const float*)d_in[2];
  float* O = (float*)d_out;
  dim3 grid(B_ * (S_ / QT));   // 4 * 64 = 256 workgroups
  dim3 block(128);             // 4 wave32 waves
  fa_fwd_bf16_wmma<<<grid, block, 0, stream>>>(Q, K, V, O);
}
